// FPNTrans_55327768707513
// MI455X (gfx1250) — matrix operand order repaired and checked by reading, not compiled, not run on hardware
//
#include <hip/hip_runtime.h>
#include <math.h>

typedef __attribute__((ext_vector_type(16))) _Float16 v16h;
typedef __attribute__((ext_vector_type(8)))  float    v8f;

#define NKEEP 196

// ---------------------------------------------------------------------------
// Implicit-GEMM conv (stride 1, "same" size) using WMMA f32_16x16x32_f16.
// One wave computes a 16(Cout) x 64(spatial) tile: A fragment loaded once per
// K-step, im2col decode done once per k, 4 WMMAs issued back-to-back.
// Epilogue: optional bias, optional BN (scale/shift), optional ReLU.
// Requires Ktot % 32 == 0 (true for every conv here); spatial tail guarded.
// ---------------------------------------------------------------------------
__global__ void conv_wmma_kernel(const float* __restrict__ X,
                                 const float* __restrict__ Wt,
                                 float* __restrict__ Y,
                                 const float* __restrict__ bias,
                                 const float* __restrict__ bnscale,
                                 const float* __restrict__ bnshift,
                                 int Cout, int H, int W, int KH, int pad, int dil,
                                 long xBatch, long yBatch, int relu, int Ktot)
{
    const int lane = threadIdx.x & 31;
    const int g    = lane >> 4;
    const int l16  = lane & 15;
    const int HW   = H * W;
    const float* Xb = X + (long)blockIdx.z * xBatch;
    float*       Yb = Y + (long)blockIdx.z * yBatch;
    const int co     = blockIdx.y * 16 + l16;   // A row owned by this lane
    const int spbase = blockIdx.x * 64;

    int oh_[4], ow_[4];
#pragma unroll
    for (int s = 0; s < 4; ++s) {
        int sp = spbase + s * 16 + l16;
        oh_[s] = sp / W;
        ow_[s] = sp - oh_[s] * W;
    }

    v8f acc[4];
#pragma unroll
    for (int s = 0; s < 4; ++s) acc[s] = (v8f){0.f,0.f,0.f,0.f,0.f,0.f,0.f,0.f};

    const float* wrow = Wt + (long)co * Ktot;
    for (int k0 = 0; k0 < Ktot; k0 += 32) {
        if (co < Cout && k0 + 32 < Ktot)
            __builtin_prefetch(wrow + k0 + 32, 0, 0);   // global_prefetch_b8
        // ---- A fragment: weights, row co.  OIHW is row-major over k. ----
        v16h a;
#pragma unroll
        for (int j = 0; j < 4; ++j) {
            int k1 = k0 + 8*g + 2*j;
            float x0 = 0.f, x1 = 0.f, x2 = 0.f, x3 = 0.f;
            if (co < Cout) {
                x0 = wrow[k1];      x1 = wrow[k1 + 1];
                x2 = wrow[k1 + 16]; x3 = wrow[k1 + 17];
            }
            a[2*j]     = (_Float16)x0;  a[2*j + 1] = (_Float16)x1;
            a[2*j + 8] = (_Float16)x2;  a[2*j + 9] = (_Float16)x3;
        }
        // ---- 4 B fragments: im2col gather, decode (ci,kh,kw) once per k ----
        v16h bfr[4];
#pragma unroll
        for (int j = 0; j < 16; ++j) {
            int kk = k0 + 8*g + j + (j >= 8 ? 8 : 0);
            int ci, kh, kw;
            if (KH == 1) { ci = kk; kh = 0; kw = 0; }
            else         { ci = kk / 9; int r = kk - ci * 9; kh = r / 3; kw = r - kh * 3; }
            const float* Xc = Xb + (long)ci * HW;
            int dh = -pad + dil * kh;
            int dw = -pad + dil * kw;
#pragma unroll
            for (int s = 0; s < 4; ++s) {
                int ih = oh_[s] + dh;
                int iw = ow_[s] + dw;
                float v = 0.f;
                if (ih >= 0 && ih < H && iw >= 0 && iw < W)
                    v = Xc[ih * W + iw];
                bfr[s][j] = (_Float16)v;
            }
        }
#pragma unroll
        for (int s = 0; s < 4; ++s)
            acc[s] = __builtin_amdgcn_wmma_f32_16x16x32_f16(false, a, false, bfr[s],
                                                            (short)0, acc[s], false, false);
    }
    // ---- D layout: lane l: N=l&15 ; VGPR r -> M = 8*(l>>4) + r ----
#pragma unroll
    for (int s = 0; s < 4; ++s) {
        int sp = spbase + s * 16 + l16;
        if (sp < HW) {
#pragma unroll
            for (int r = 0; r < 8; ++r) {
                int c = blockIdx.y * 16 + 8*g + r;
                if (c < Cout) {
                    float v = acc[s][r];
                    if (bias)    v += bias[c];
                    if (bnscale) v = v * bnscale[c] + bnshift[c];
                    if (relu)    v = fmaxf(v, 0.f);
                    Yb[(long)c * HW + sp] = v;
                }
            }
        }
    }
}

// ---------------------------------------------------------------------------
// Generic WMMA GEMM:  C = alpha * A(MxK) * B + bias
//   transB=1: B is [N,K] row-major (i.e. X @ W^T);  transB=0: B is [K,N].
// One wave computes a 16(M) x 64(N) tile (4 accumulators, A loaded once).
// Batch via gridDim.z with a 2-level stride decomposition: z = z0*zdiv + z1.
// Full bounds checks on M/N/K (handles M=392, N=196, K=196 tails).
// ---------------------------------------------------------------------------
__global__ void gemm_wmma_kernel(const float* __restrict__ A,
                                 const float* __restrict__ B,
                                 float* __restrict__ C,
                                 const float* __restrict__ bias,
                                 int M, int N, int K, int lda, int ldb, int ldc,
                                 int zdiv, long sA0, long sA1, long sB0, long sB1,
                                 long sC0, long sC1, int transB, float alpha)
{
    const int lane = threadIdx.x & 31;
    const int g    = lane >> 4;
    const int l16  = lane & 15;
    const int z  = blockIdx.z;
    const int z0 = z / zdiv;
    const int z1 = z - z0 * zdiv;
    const float* Ab = A + z0 * sA0 + z1 * sA1;
    const float* Bb = B + z0 * sB0 + z1 * sB1;
    float*       Cb = C + z0 * sC0 + z1 * sC1;
    const int m     = blockIdx.y * 16 + l16;
    const int nbase = blockIdx.x * 64;

    v8f acc[4];
#pragma unroll
    for (int s = 0; s < 4; ++s) acc[s] = (v8f){0.f,0.f,0.f,0.f,0.f,0.f,0.f,0.f};

    for (int k0 = 0; k0 < K; k0 += 32) {
        v16h a;
#pragma unroll
        for (int j = 0; j < 4; ++j) {
            int k1 = k0 + 8*g + 2*j;
            int k2 = k1 + 16;
            float x0 = 0.f, x1 = 0.f, x2 = 0.f, x3 = 0.f;
            if (m < M) {
                const float* ar = Ab + (long)m * lda;
                if (k1     < K) x0 = ar[k1];
                if (k1 + 1 < K) x1 = ar[k1 + 1];
                if (k2     < K) x2 = ar[k2];
                if (k2 + 1 < K) x3 = ar[k2 + 1];
            }
            a[2*j]     = (_Float16)x0;  a[2*j + 1] = (_Float16)x1;
            a[2*j + 8] = (_Float16)x2;  a[2*j + 9] = (_Float16)x3;
        }
        v16h bfr[4];
#pragma unroll
        for (int j = 0; j < 16; ++j) {
            int kk = k0 + 8*g + j + (j >= 8 ? 8 : 0);
#pragma unroll
            for (int s = 0; s < 4; ++s) {
                int n = nbase + s * 16 + l16;
                float v = 0.f;
                if (n < N && kk < K)
                    v = transB ? Bb[(long)n * ldb + kk] : Bb[(long)kk * ldb + n];
                bfr[s][j] = (_Float16)v;
            }
        }
#pragma unroll
        for (int s = 0; s < 4; ++s)
            acc[s] = __builtin_amdgcn_wmma_f32_16x16x32_f16(false, a, false, bfr[s],
                                                            (short)0, acc[s], false, false);
    }
#pragma unroll
    for (int s = 0; s < 4; ++s) {
        int n = nbase + s * 16 + l16;
        if (n < N) {
            float bv = bias ? bias[n] : 0.f;
#pragma unroll
            for (int r = 0; r < 8; ++r) {
                int mm = blockIdx.y * 16 + 8*g + r;
                if (mm < M) Cb[(long)mm * ldc + n] = alpha * acc[s][r] + bv;
            }
        }
    }
}

// ------------------------------ LayerNorm (D=256) ---------------------------
__global__ void ln_kernel(const float* __restrict__ x, const float* __restrict__ gw,
                          const float* __restrict__ bw, float* __restrict__ y)
{
    __shared__ float red[256];
    const int row = blockIdx.x, d = threadIdx.x;
    float v = x[(long)row * 256 + d];
    red[d] = v; __syncthreads();
    for (int s = 128; s > 0; s >>= 1) { if (d < s) red[d] += red[d + s]; __syncthreads(); }
    float mu = red[0] * (1.f / 256.f);
    __syncthreads();
    float c = v - mu;
    red[d] = c * c; __syncthreads();
    for (int s = 128; s > 0; s >>= 1) { if (d < s) red[d] += red[d + s]; __syncthreads(); }
    float var = red[0] * (1.f / 256.f);
    y[(long)row * 256 + d] = c * rsqrtf(var + 1e-5f) * gw[d] + bw[d];
}

// --------------------------- Softmax over rows of 196 -----------------------
__global__ void softmax196_kernel(float* __restrict__ x)
{
    __shared__ float red[256];
    const int row = blockIdx.x, d = threadIdx.x;
    float v = (d < 196) ? x[(long)row * 196 + d] : -3.0e38f;
    red[d] = v; __syncthreads();
    for (int s = 128; s > 0; s >>= 1) { if (d < s) red[d] = fmaxf(red[d], red[d + s]); __syncthreads(); }
    float mx = red[0]; __syncthreads();
    float e = (d < 196) ? expf(v - mx) : 0.f;
    red[d] = e; __syncthreads();
    for (int s = 128; s > 0; s >>= 1) { if (d < s) red[d] += red[d + s]; __syncthreads(); }
    float inv = 1.f / red[0];
    if (d < 196) x[(long)row * 196 + d] = e * inv;
}

// ----------------------------- elementwise glue ------------------------------
__global__ void residual_kernel(float* __restrict__ x, const float* __restrict__ t,
                                const float* __restrict__ gam, int total)
{
    int i = blockIdx.x * blockDim.x + threadIdx.x;
    if (i < total) x[i] += gam[i & 255] * t[i];
}

__global__ void silu_kernel(float* __restrict__ x, int total)
{
    int i = blockIdx.x * blockDim.x + threadIdx.x;
    if (i < total) { float v = x[i]; x[i] = v / (1.f + expf(-1.702f * v)); }
}

// tok[n][t][d] = in5[n][d][p] + pos[d][p], p = idx[n][t]
__global__ void tok_gather_kernel(const float* __restrict__ in5, const float* __restrict__ pos,
                                  const int* __restrict__ idx, float* __restrict__ tok)
{
    int i = blockIdx.x * blockDim.x + threadIdx.x;
    if (i >= 2 * NKEEP * 256) return;
    int d = i & 255;
    int t = (i >> 8) % NKEEP;
    int n = i / (NKEEP * 256);
    int p = idx[n * NKEEP + t];
    tok[i] = in5[((long)n * 256 + d) * 784 + p] + pos[(long)d * 784 + p];
}

__global__ void tok_scatter_kernel(float* __restrict__ in5, const float* __restrict__ tok,
                                   const int* __restrict__ idx)
{
    int i = blockIdx.x * blockDim.x + threadIdx.x;
    if (i >= 2 * NKEEP * 256) return;
    int d = i & 255;
    int t = (i >> 8) % NKEEP;
    int n = i / (NKEEP * 256);
    int p = idx[n * NKEEP + t];
    in5[((long)n * 256 + d) * 784 + p] = tok[i];
}

// dst[plane][h][w] += src[plane][h/2][w/2]
__global__ void upadd_kernel(float* __restrict__ dst, const float* __restrict__ src,
                             int H, int W, long total)
{
    long i = (long)blockIdx.x * blockDim.x + threadIdx.x;
    if (i >= total) return;
    int w = (int)(i % W);
    long r = i / W;
    int h = (int)(r % H);
    long plane = r / H;
    int Hs = H >> 1, Ws = W >> 1;
    dst[i] += src[(plane * Hs + (h >> 1)) * Ws + (w >> 1)];
}

// feat[2][256][224][224] channel slice [coff..coff+64) = up(src [2][64][224/s][224/s], s)
__global__ void upcopy_kernel(float* __restrict__ feat, const float* __restrict__ src,
                              int sfac, int coff)
{
    long i = (long)blockIdx.x * blockDim.x + threadIdx.x;
    if (i >= (long)2 * 64 * 224 * 224) return;
    int w = (int)(i % 224);
    long r = i / 224;
    int h = (int)(r % 224); r /= 224;
    int c = (int)(r % 64);
    int n = (int)(r / 64);
    int Hs = 224 / sfac;
    feat[(((long)n * 256 + coff + c) * 224 + h) * 224 + w] =
        src[(((long)n * 64 + c) * Hs + h / sfac) * Hs + w / sfac];
}

__global__ void bnprep_kernel(const float* __restrict__ g, const float* __restrict__ b,
                              const float* __restrict__ m, const float* __restrict__ v,
                              float* __restrict__ scale, float* __restrict__ shift)
{
    int c = threadIdx.x;
    float s = g[c] * rsqrtf(v[c] + 1e-5f);
    scale[c] = s;
    shift[c] = b[c] - m[c] * s;
}

// sigmoid+clip -> ksc -> keep_score2 output, gumbel softmax -> gscore, ghard
__global__ void ks_post_kernel(const float* __restrict__ s3, const float* __restrict__ gum,
                               float* __restrict__ ks2_out, float* __restrict__ gscore,
                               float* __restrict__ ghard)
{
    int i = blockIdx.x * blockDim.x + threadIdx.x;
    if (i >= 2 * 50176) return;
    int n = i / 50176;
    int p = i - n * 50176;
    float sv = 1.f / (1.f + expf(-s3[i]));
    sv = fminf(fmaxf(sv, 0.f), 1.f);
    float k0 = sv + 1e-5f;
    float k1 = (1.f - sv) + 1e-5f;
    ks2_out[i] = k0;
    float z0 = logf(k0) + gum[(long)n * 100352 + p];
    float z1 = logf(k1) + gum[(long)n * 100352 + 50176 + p];
    float mx = fmaxf(z0, z1);
    float e0 = expf(z0 - mx), e1 = expf(z1 - mx);
    float inv = 1.f / (e0 + e1);
    float s0 = e0 * inv, s1v = e1 * inv;
    gscore[i] = s0;
    float oh = (s0 >= s1v) ? 1.f : 0.f;
    ghard[i] = (oh - s0) + s0;   // replicate reference float op order
}

// exact rank: keep[i] = (#{v_j > v_i} + #{j<i : v_j == v_i}) < NKEEP
__global__ void topk_rank_kernel(const float* __restrict__ vals, int* __restrict__ keep, int N)
{
    __shared__ float tile[256];
    const int b = blockIdx.y;
    const float* v = vals + (long)b * N;
    const int i = blockIdx.x * 256 + threadIdx.x;
    float vi = (i < N) ? v[i] : -3.0e38f;
    int rank = 0;
    for (int t0 = 0; t0 < N; t0 += 256) {
        int j = t0 + threadIdx.x;
        tile[threadIdx.x] = (j < N) ? v[j] : -3.0e38f;
        __syncthreads();
        int lim = N - t0; if (lim > 256) lim = 256;
        for (int jj = 0; jj < lim; ++jj) {
            float vj = tile[jj];
            int j2 = t0 + jj;
            if (vj > vi || (vj == vi && j2 < i)) rank++;
        }
        __syncthreads();
    }
    if (i < N) keep[(long)b * N + i] = (rank < NKEEP) ? 1 : 0;
}

// compact kept indices in ascending index order -> sorted top-k indices
__global__ void topk_compact_kernel(const int* __restrict__ keep, int* __restrict__ idx, int N)
{
    __shared__ int cnt[1024];
    const int b = blockIdx.x, t = threadIdx.x;
    const int* kp = keep + (long)b * N;
    int chunk = (N + 1023) / 1024;
    int beg = t * chunk;
    int end = beg + chunk; if (end > N) end = N;
    int c = 0;
    for (int i = beg; i < end; ++i) c += kp[i];
    cnt[t] = c; __syncthreads();
    if (t == 0) { int acc = 0; for (int i = 0; i < 1024; ++i) { int x = cnt[i]; cnt[i] = acc; acc += x; } }
    __syncthreads();
    int pos = cnt[t];
    for (int i = beg; i < end; ++i) {
        if (kp[i]) { if (pos < NKEEP) idx[(long)b * NKEEP + pos] = i; pos++; }
    }
}

// out[t][n][d] = (feat[n][d][p] + pos2[d][p]) * ghard[n][p], p = idx2[n][t]
__global__ void sample_gather_kernel(const float* __restrict__ feat, const float* __restrict__ pos2,
                                     const float* __restrict__ ghard, const int* __restrict__ idx2,
                                     float* __restrict__ out)
{
    int i = blockIdx.x * blockDim.x + threadIdx.x;
    if (i >= NKEEP * 2 * 256) return;
    int d = i & 255;
    int n = (i >> 8) & 1;
    int t = i >> 9;
    int p = idx2[n * NKEEP + t];
    float v = feat[((long)n * 256 + d) * 50176 + p] + pos2[(long)d * 50176 + p];
    out[i] = v * ghard[(long)n * 50176 + p];
}

// ---------------------------------------------------------------------------
extern "C" void kernel_launch(void* const* d_in, const int* in_sizes, int n_in,
                              void* d_out, int out_size, void* d_ws, size_t ws_size,
                              hipStream_t stream)
{
    (void)in_sizes; (void)n_in; (void)out_size; (void)ws_size;
    const float* c2        = (const float*)d_in[0];
    const float* c3        = (const float*)d_in[1];
    const float* c4        = (const float*)d_in[2];
    const float* c5        = (const float*)d_in[3];
    const float* in5_w     = (const float*)d_in[4];
    const float* in4_w     = (const float*)d_in[5];
    const float* in3_w     = (const float*)d_in[6];
    const float* in2_w     = (const float*)d_in[7];
    const float* out5_w    = (const float*)d_in[8];
    const float* out4_w    = (const float*)d_in[9];
    const float* out3_w    = (const float*)d_in[10];
    const float* out2_w    = (const float*)d_in[11];
    const float* gs1_w     = (const float*)d_in[12];
    const float* gs1_b     = (const float*)d_in[13];
    const float* pos1      = (const float*)d_in[14];
    const float* pos2      = (const float*)d_in[15];
    const float* vln1_g    = (const float*)d_in[16];
    const float* vln1_b    = (const float*)d_in[17];
    const float* vqkv_w    = (const float*)d_in[18];
    const float* vqkv_b    = (const float*)d_in[19];
    const float* vout_w    = (const float*)d_in[20];
    const float* vout_b    = (const float*)d_in[21];
    const float* vln2_g    = (const float*)d_in[22];
    const float* vln2_b    = (const float*)d_in[23];
    const float* vfc_w     = (const float*)d_in[24];
    const float* vfc_b     = (const float*)d_in[25];
    const float* vproj_w   = (const float*)d_in[26];
    const float* vproj_b   = (const float*)d_in[27];
    const float* vg1       = (const float*)d_in[28];
    const float* vg2       = (const float*)d_in[29];
    const float* ks_w1     = (const float*)d_in[30];
    const float* bn1_g     = (const float*)d_in[31];
    const float* bn1_b     = (const float*)d_in[32];
    const float* bn1_m     = (const float*)d_in[33];
    const float* bn1_v     = (const float*)d_in[34];
    const float* ks_w2     = (const float*)d_in[35];
    const float* bn2_g     = (const float*)d_in[36];
    const float* bn2_b     = (const float*)d_in[37];
    const float* bn2_m     = (const float*)d_in[38];
    const float* bn2_v     = (const float*)d_in[39];
    const float* ks_w3     = (const float*)d_in[40];
    const float* ks_b3     = (const float*)d_in[41];
    const float* rand1     = (const float*)d_in[42];
    const float* gumbel2   = (const float*)d_in[43];

    // ------------------- workspace layout (floats) -------------------
    float* ws = (float*)d_ws;
    size_t off = 0;
    auto alloc = [&](size_t n) { float* p = ws + off; off += (n + 63) & ~(size_t)63; return p; };
    float* IN5  = alloc(401408);     // [2,256,28,28]
    float* IN5B = alloc(401408);
    float* IN4  = alloc(1605632);    // [2,256,56,56]
    float* IN3  = alloc(6422528);    // [2,256,112,112]
    float* IN2  = alloc(25690112);   // [2,256,224,224]  (later aliased as s1)
    float* S2   = alloc(25690112);
    float* CTMP = alloc(1605632);    // out-conv temp (max [2,64,112,112])
    float* TOK  = alloc(100352);     // [2,196,256]
    float* HB   = alloc(100352);
    float* QKV  = alloc(301056);     // [2,196,768]
    float* SC   = alloc(307328);     // [2,4,196,196]
    float* OM   = alloc(100352);     // merged attn out [2,196,256]
    float* FCB  = alloc(200704);     // [2,196,512]
    float* S3   = alloc(100352);     // [2,1,224,224]
    float* GSC  = alloc(100352);
    float* GHD  = alloc(100352);
    float* BNS1 = alloc(256); float* BNB1 = alloc(256);
    float* BNS2 = alloc(256); float* BNB2 = alloc(256);
    int*   KEEPF = (int*)alloc(100352);
    int*   IDX1  = (int*)alloc(512);
    int*   IDX2  = (int*)alloc(512);
    float* S1 = IN2;                 // reuse after feat_ms finished

    float* FEAT   = (float*)d_out;                 // [2,256,224,224]
    float* SAMP   = FEAT + (long)25690112;         // [196,2,256]
    float* KS2OUT = FEAT + (long)25690112 + 100352;

    // ---------------- BN fold ----------------
    bnprep_kernel<<<1, 256, 0, stream>>>(bn1_g, bn1_b, bn1_m, bn1_v, BNS1, BNB1);
    bnprep_kernel<<<1, 256, 0, stream>>>(bn2_g, bn2_b, bn2_m, bn2_v, BNS2, BNB2);

    // ---------------- lateral 1x1 convs (x-grid = ceil(HW/64)) ----------------
    conv_wmma_kernel<<<dim3(13, 16, 2),  32, 0, stream>>>(c5, in5_w, IN5, nullptr, nullptr, nullptr,
        256, 28, 28, 1, 0, 1, (long)768 * 784, (long)256 * 784, 0, 768);
    conv_wmma_kernel<<<dim3(49, 16, 2),  32, 0, stream>>>(c4, in4_w, IN4, nullptr, nullptr, nullptr,
        256, 56, 56, 1, 0, 1, (long)384 * 3136, (long)256 * 3136, 0, 384);
    conv_wmma_kernel<<<dim3(196, 16, 2), 32, 0, stream>>>(c3, in3_w, IN3, nullptr, nullptr, nullptr,
        256, 112, 112, 1, 0, 1, (long)192 * 12544, (long)256 * 12544, 0, 192);
    conv_wmma_kernel<<<dim3(784, 16, 2), 32, 0, stream>>>(c2, in2_w, IN2, nullptr, nullptr, nullptr,
        256, 224, 224, 1, 0, 1, (long)96 * 50176, (long)256 * 50176, 0, 96);

    // ---------------- token selection (idx1) + gather ----------------
    topk_rank_kernel<<<dim3(4, 2), 256, 0, stream>>>(rand1, KEEPF, 784);
    topk_compact_kernel<<<2, 1024, 0, stream>>>(KEEPF, IDX1, 784);
    tok_gather_kernel<<<392, 256, 0, stream>>>(IN5, pos1, IDX1, TOK);

    // ---------------- 2 transformer blocks on [2,196,256] ----------------
    for (int l = 0; l < 2; ++l) {
        ln_kernel<<<392, 256, 0, stream>>>(TOK, vln1_g + l * 256, vln1_b + l * 256, HB);
        // qkv = h @ qkv_w^T + b : M=392 N=768 K=256
        gemm_wmma_kernel<<<dim3(12, 25, 1), 32, 0, stream>>>(HB, vqkv_w + (long)l * 768 * 256, QKV,
            vqkv_b + l * 768, 392, 768, 256, 256, 256, 768,
            1, 0, 0, 0, 0, 0, 0, 1, 1.0f);
        // scores = q k^T / 8 per (n,h): z = n*4+h
        gemm_wmma_kernel<<<dim3(4, 13, 8), 32, 0, stream>>>(QKV, QKV + 256, SC, nullptr,
            196, 196, 64, 768, 768, 196,
            4, 150528, 64, 150528, 64, 153664, 38416, 1, 0.125f);
        softmax196_kernel<<<1568, 256, 0, stream>>>(SC);
        // o = a @ v, written directly into merged [n,l,h*64+d] layout
        gemm_wmma_kernel<<<dim3(1, 13, 8), 32, 0, stream>>>(SC, QKV + 512, OM, nullptr,
            196, 64, 196, 196, 768, 256,
            4, 153664, 38416, 150528, 64, 50176, 64, 0, 1.0f);
        // out proj
        gemm_wmma_kernel<<<dim3(4, 25, 1), 32, 0, stream>>>(OM, vout_w + (long)l * 256 * 256, HB,
            vout_b + l * 256, 392, 256, 256, 256, 256, 256,
            1, 0, 0, 0, 0, 0, 0, 1, 1.0f);
        residual_kernel<<<392, 256, 0, stream>>>(TOK, HB, vg1 + l * 256, 100352);
        // MLP
        ln_kernel<<<392, 256, 0, stream>>>(TOK, vln2_g + l * 256, vln2_b + l * 256, HB);
        gemm_wmma_kernel<<<dim3(8, 25, 1), 32, 0, stream>>>(HB, vfc_w + (long)l * 512 * 256, FCB,
            vfc_b + l * 512, 392, 512, 256, 256, 256, 512,
            1, 0, 0, 0, 0, 0, 0, 1, 1.0f);
        silu_kernel<<<784, 256, 0, stream>>>(FCB, 200704);
        gemm_wmma_kernel<<<dim3(4, 25, 1), 32, 0, stream>>>(FCB, vproj_w + (long)l * 256 * 512, HB,
            vproj_b + l * 256, 392, 256, 512, 512, 512, 256,
            1, 0, 0, 0, 0, 0, 0, 1, 1.0f);
        residual_kernel<<<392, 256, 0, stream>>>(TOK, HB, vg2 + l * 256, 100352);
    }

    // scatter tokens back into in5 (no pos-emb in the base, matching reference)
    tok_scatter_kernel<<<392, 256, 0, stream>>>(IN5, TOK, IDX1);

    // gs1_diff 3x3 conv with bias
    conv_wmma_kernel<<<dim3(13, 16, 2), 32, 0, stream>>>(IN5, gs1_w, IN5B, gs1_b, nullptr, nullptr,
        256, 28, 28, 3, 1, 1, (long)256 * 784, (long)256 * 784, 0, 2304);

    // FPN top-down accumulate
    upadd_kernel<<<(2 * 256 * 3136 + 255) / 256, 256, 0, stream>>>(IN4, IN5B, 56, 56, (long)2 * 256 * 3136);
    upadd_kernel<<<(2 * 256 * 12544 + 255) / 256, 256, 0, stream>>>(IN3, IN4, 112, 112, (long)2 * 256 * 12544);
    upadd_kernel<<<(2 * 256 * 50176 + 255) / 256, 256, 0, stream>>>(IN2, IN3, 224, 224, (long)2 * 256 * 50176);

    // output heads -> feat_ms slices (p5,p4,p3,p2)
    conv_wmma_kernel<<<dim3(13, 4, 2), 32, 0, stream>>>(IN5B, out5_w, CTMP, nullptr, nullptr, nullptr,
        64, 28, 28, 3, 1, 1, (long)256 * 784, (long)64 * 784, 0, 2304);
    upcopy_kernel<<<(2 * 64 * 50176 + 255) / 256, 256, 0, stream>>>(FEAT, CTMP, 8, 0);
    conv_wmma_kernel<<<dim3(49, 4, 2), 32, 0, stream>>>(IN4, out4_w, CTMP, nullptr, nullptr, nullptr,
        64, 56, 56, 3, 1, 1, (long)256 * 3136, (long)64 * 3136, 0, 2304);
    upcopy_kernel<<<(2 * 64 * 50176 + 255) / 256, 256, 0, stream>>>(FEAT, CTMP, 4, 64);
    conv_wmma_kernel<<<dim3(196, 4, 2), 32, 0, stream>>>(IN3, out3_w, CTMP, nullptr, nullptr, nullptr,
        64, 112, 112, 3, 1, 1, (long)256 * 12544, (long)64 * 12544, 0, 2304);
    upcopy_kernel<<<(2 * 64 * 50176 + 255) / 256, 256, 0, stream>>>(FEAT, CTMP, 2, 128);
    conv_wmma_kernel<<<dim3(784, 4, 2), 32, 0, stream>>>(IN2, out2_w, FEAT + (long)192 * 50176,
        nullptr, nullptr, nullptr,
        64, 224, 224, 3, 1, 1, (long)256 * 50176, (long)256 * 50176, 0, 2304);

    // keep-score branch (compute bound: WMMA convs with fused BN+ReLU)
    conv_wmma_kernel<<<dim3(784, 16, 2), 32, 0, stream>>>(FEAT, ks_w1, S1, nullptr, BNS1, BNB1,
        256, 224, 224, 3, 3, 3, (long)256 * 50176, (long)256 * 50176, 1, 2304);
    conv_wmma_kernel<<<dim3(784, 16, 2), 32, 0, stream>>>(S1, ks_w2, S2, nullptr, BNS2, BNB2,
        256, 224, 224, 3, 2, 2, (long)256 * 50176, (long)256 * 50176, 1, 2304);
    conv_wmma_kernel<<<dim3(784, 1, 2), 32, 0, stream>>>(S2, ks_w3, S3, ks_b3, nullptr, nullptr,
        1, 224, 224, 3, 1, 1, (long)256 * 50176, (long)50176, 0, 2304);

    ks_post_kernel<<<392, 256, 0, stream>>>(S3, gumbel2, KS2OUT, GSC, GHD);

    // idx2 selection + final sampled gather
    topk_rank_kernel<<<dim3(196, 2), 256, 0, stream>>>(GSC, KEEPF, 50176);
    topk_compact_kernel<<<2, 1024, 0, stream>>>(KEEPF, IDX2, 50176);
    sample_gather_kernel<<<392, 256, 0, stream>>>(FEAT, pos2, GHD, IDX2, SAMP);
}
